// QuantumKernelMethod_65481071397145
// MI455X (gfx1250) — matrix operand order repaired and checked by reading, not compiled, not run on hardware
//
#include <hip/hip_runtime.h>

// ---------------------------------------------------------------------------
// RBF kernel matrix: out[b,c] = exp(-gamma * max(0, |x_b|^2 + |y_c|^2 - 2 x_b.y_c))
// B = C = 8192, D = 256, fp32 in / fp32 out.
// Strategy: f16 WMMA GEMM (fp32 accumulate) + fused norm/exp epilogue.
// Roofline: 34.4 GFLOP f16 WMMA (~7us at ~5PF dense) vs 272MB HBM (~12us)
// -> store-bandwidth bound; NT stores keep the 256MB output stream out of L2
// where the 16MB of x/y operand data stays resident.
// ---------------------------------------------------------------------------

#define BDIM   8192   // rows of x (B)
#define CDIM   8192   // rows of y (C)
#define DDIM   256    // feature dim (K total)
#define TILE_M 128
#define TILE_N 128
#define TILE_K 32
#define NKSTEP (DDIM / TILE_K)   // 8
#define LDH    40                // padded halves per LDS row (80 B, 16B-aligned)

typedef __attribute__((ext_vector_type(16))) _Float16 v16h;
typedef __attribute__((ext_vector_type(8)))  _Float16 v8h;
typedef __attribute__((ext_vector_type(2)))  __fp16   h2raw;  // cvt_pkrtz result type
typedef __attribute__((ext_vector_type(8)))  float    v8f;

// ---------------------------------------------------------------------------
// Row-norm kernel: one wave32 per row, 256 floats/row -> 2 float4 per lane.
// ---------------------------------------------------------------------------
__global__ void __launch_bounds__(256)
rbf_rownorm_kernel(const float* __restrict__ src, float* __restrict__ dst) {
    const int wave = (blockIdx.x * blockDim.x + threadIdx.x) >> 5;
    const int lane = threadIdx.x & 31;
    const float4* p = (const float4*)(src + (size_t)wave * DDIM);
    float4 a = p[lane];
    float4 b = p[lane + 32];
    float s = a.x*a.x + a.y*a.y + a.z*a.z + a.w*a.w
            + b.x*b.x + b.y*b.y + b.z*b.z + b.w*b.w;
    #pragma unroll
    for (int off = 16; off > 0; off >>= 1)
        s += __shfl_xor(s, off, 32);
    if (lane == 0) dst[wave] = s;
}

// ---------------------------------------------------------------------------
// WMMA fragment loads from LDS (row-major f16 tiles, row pitch LDH halves).
// A 16x32 f16 layout: lane = M (mod 16); lanes<16 hold K {0..7,16..23},
// lanes>=16 hold K {8..15,24..31}  -> two 16B ds_load_b128 per lane.
// ---------------------------------------------------------------------------
__device__ __forceinline__ v16h load_fragA(const _Float16* base, int lane) {
    const int m = lane & 15;
    const _Float16* p = base + m * LDH + ((lane >> 4) << 3); // +0 or +8 halves
    union { v16h v; struct { v8h lo, hi; } s; } u;
    u.s.lo = *(const v8h*)p;          // K chunk 0 (8 halves)
    u.s.hi = *(const v8h*)(p + 16);   // K chunk +16
    return u.v;
}

// B 32x16 f16 layout: lane = N (mod 16); lanes<16 hold K {0..7,16..23},
// lanes>=16 hold K {8..15,24..31} -> two 16B loads per lane.
__device__ __forceinline__ v16h load_fragB(const _Float16* base, int lane) {
    const int n = lane & 15;
    const _Float16* p = base + n * LDH + ((lane >> 4) << 3); // +0 or +8 halves
    union { v16h v; struct { v8h lo, hi; } s; } u;
    u.s.lo = *(const v8h*)p;
    u.s.hi = *(const v8h*)(p + 16);
    return u.v;
}

// ---------------------------------------------------------------------------
// Main fused kernel. Grid (64,64), block 256 (8 waves). Workgroup tile
// 128x128; wave (wm in 0..3, wn in 0..1) computes rows wm*32..+32,
// cols wn*64..+64 as 2x4 WMMA 16x16 tiles.
// ---------------------------------------------------------------------------
__global__ void __launch_bounds__(256)
rbf_wmma_kernel(const float* __restrict__ x, const float* __restrict__ y,
                const float* __restrict__ gamma_p,
                const float* __restrict__ nx, const float* __restrict__ ny,
                float* __restrict__ out) {
    __shared__ __align__(16) _Float16 sA[2][TILE_M * LDH];
    __shared__ __align__(16) _Float16 sB[2][TILE_N * LDH];

    const int tid  = threadIdx.x;
    const int lane = tid & 31;
    const int wave = tid >> 5;
    const int wm   = wave & 3;   // 4 waves along M
    const int wn   = wave >> 2;  // 2 waves along N
    const int row0 = blockIdx.y * TILE_M;
    const int col0 = blockIdx.x * TILE_N;

    // Global loaders: threads 0..127 stream one x row each, 128..255 one y row.
    const int ldrow = tid & 127;
    const float* grow = (tid < 128)
        ? (x + (size_t)(row0 + ldrow) * DDIM)
        : (y + (size_t)(col0 + ldrow) * DDIM);
    _Float16* lrow0 = ((tid < 128) ? &sA[0][0] : &sB[0][0]) + ldrow * LDH;
    const int lbufstride = TILE_M * LDH;

    float4 rbuf[8];
    #pragma unroll
    for (int i = 0; i < 8; ++i)          // prologue: K-slab 0 (32 floats/row)
        rbuf[i] = ((const float4*)grow)[i];

    v8f acc[2][4] = {};

    for (int kt = 0; kt < NKSTEP; ++kt) {
        const int buf = kt & 1;
        // Convert fp32 -> f16 pairs (v_cvt_pkrtz), 16B LDS stores.
        _Float16* dst = lrow0 + buf * lbufstride;
        #pragma unroll
        for (int i = 0; i < 4; ++i) {
            union { v8h v; h2raw p[4]; } u;
            u.p[0] = __builtin_amdgcn_cvt_pkrtz(rbuf[2*i].x,   rbuf[2*i].y);
            u.p[1] = __builtin_amdgcn_cvt_pkrtz(rbuf[2*i].z,   rbuf[2*i].w);
            u.p[2] = __builtin_amdgcn_cvt_pkrtz(rbuf[2*i+1].x, rbuf[2*i+1].y);
            u.p[3] = __builtin_amdgcn_cvt_pkrtz(rbuf[2*i+1].z, rbuf[2*i+1].w);
            *(v8h*)(dst + i * 8) = u.v;
        }
        // Prefetch next K-slab into registers (overlaps the WMMA burst).
        if (kt + 1 < NKSTEP) {
            const float4* gp = (const float4*)grow + (kt + 1) * 8;
            #pragma unroll
            for (int i = 0; i < 8; ++i) rbuf[i] = gp[i];
        }
        __syncthreads();   // single barrier per step; double buffer covers WAR

        v16h afrag[2], bfrag[4];
        #pragma unroll
        for (int am = 0; am < 2; ++am)
            afrag[am] = load_fragA(&sA[buf][(wm * 32 + am * 16) * LDH], lane);
        #pragma unroll
        for (int bn = 0; bn < 4; ++bn)
            bfrag[bn] = load_fragB(&sB[buf][(wn * 64 + bn * 16) * LDH], lane);
        #pragma unroll
        for (int am = 0; am < 2; ++am)
            #pragma unroll
            for (int bn = 0; bn < 4; ++bn)
                acc[am][bn] = __builtin_amdgcn_wmma_f32_16x16x32_f16(
                    false, afrag[am], false, bfrag[bn],
                    (short)0, acc[am][bn], false, false);
    }

    // Fused epilogue: sq = nx + ny - 2*dot, clamp, exp; NT stores (write-once
    // 256 MB stream must not evict the hot x/y tiles from L2).
    const float gamma = *gamma_p;
    #pragma unroll
    for (int am = 0; am < 2; ++am) {
        const int gr0 = row0 + wm * 32 + am * 16 + ((lane >> 4) << 3);
        const float4 n0 = *(const float4*)(nx + gr0);
        const float4 n1 = *(const float4*)(nx + gr0 + 4);
        const float nxa[8] = { n0.x, n0.y, n0.z, n0.w, n1.x, n1.y, n1.z, n1.w };
        #pragma unroll
        for (int bn = 0; bn < 4; ++bn) {
            const int gc = col0 + wn * 64 + bn * 16 + (lane & 15);
            const float y2 = ny[gc];
            v8f a = acc[am][bn];
            #pragma unroll
            for (int r = 0; r < 8; ++r) {
                float sq = nxa[r] + y2 - 2.0f * a[r];
                sq = fmaxf(sq, 0.0f);
                const float v = __expf(-gamma * sq);
                __builtin_nontemporal_store(v, out + (size_t)(gr0 + r) * CDIM + gc);
            }
        }
    }
}

// ---------------------------------------------------------------------------
extern "C" void kernel_launch(void* const* d_in, const int* in_sizes, int n_in,
                              void* d_out, int out_size, void* d_ws, size_t ws_size,
                              hipStream_t stream) {
    const float* x     = (const float*)d_in[0];   // (8192, 256)
    const float* y     = (const float*)d_in[1];   // (8192, 256)
    const float* gamma = (const float*)d_in[2];   // scalar
    float* out = (float*)d_out;                   // (8192, 8192)
    float* nx  = (float*)d_ws;                    // 8192 floats
    float* ny  = nx + BDIM;                       // 8192 floats (64 KB total)

    // 8 waves/block -> 1024 blocks cover 8192 rows each.
    rbf_rownorm_kernel<<<dim3(BDIM / 8), 256, 0, stream>>>(x, nx);
    rbf_rownorm_kernel<<<dim3(CDIM / 8), 256, 0, stream>>>(y, ny);

    rbf_wmma_kernel<<<dim3(CDIM / TILE_N, BDIM / TILE_M), 256, 0, stream>>>(
        x, y, gamma, nx, ny, out);
}
